// CC_AlexNet_37469294690896
// MI455X (gfx1250) — matrix operand order repaired and checked by reading, not compiled, not run on hardware
//
#include <hip/hip_runtime.h>

// ---------------------------------------------------------------------------
// CondConv-AlexNet forward, MI455X (gfx1250, wave32).
// Convs + FCs via v_wmma_f32_16x16x32_bf16 implicit GEMM, f32 accumulate.
// Conv kernel templated on KW (compile-time im2col decode) and MW
// (A-fragments per wave); K-loop split into guard-free main chunks + guarded
// tail, with an interior fast path that avoids per-element exec predication.
// ---------------------------------------------------------------------------

typedef __attribute__((ext_vector_type(16))) __bf16 v16bf;
typedef __attribute__((ext_vector_type(8)))  float  v8f;
typedef unsigned short u16;
typedef unsigned int   u32;

#define LDSK 40   // LDS B-tile row stride in bf16 elems (80B: 16B aligned, bank-friendly)

__device__ __forceinline__ float bf2f(u16 u) {
    u32 i = ((u32)u) << 16; float f; __builtin_memcpy(&f, &i, 4); return f;
}
__device__ __forceinline__ u16 f2bf(float f) {
    u32 i; __builtin_memcpy(&i, &f, 4);
    return (u16)((i + 0x7FFFu + ((i >> 16) & 1u)) >> 16);   // RNE
}

// ---------------------------------------------------------------------------
// fp32 -> bf16 convert
__global__ void k_f32_to_bf16(const float* __restrict__ in, u16* __restrict__ out, long long n) {
    long long i = (long long)blockIdx.x * blockDim.x + threadIdx.x;
    if (i < n) out[i] = f2bf(in[i]);
}

// ---------------------------------------------------------------------------
// Global average pool over H*W: pooled[b*C+c]
__global__ void k_mean_hw(const u16* __restrict__ act, float* __restrict__ pooled,
                          int C, int HW) {
    int bc = blockIdx.x;
    const u16* p = act + (long long)bc * HW;
    float s = 0.f;
    for (int i = threadIdx.x; i < HW; i += blockDim.x) s += bf2f(p[i]);
    __shared__ float red[256];
    red[threadIdx.x] = s; __syncthreads();
    for (int off = blockDim.x >> 1; off > 0; off >>= 1) {
        if (threadIdx.x < (unsigned)off) red[threadIdx.x] += red[threadIdx.x + off];
        __syncthreads();
    }
    if (threadIdx.x == 0) pooled[bc] = red[0] / (float)HW;
}

// ---------------------------------------------------------------------------
// Routing: r[b,k] = sigmoid(pooled@rw + rb); bb[b,cout] = r @ bexp
__global__ void k_routing(const float* __restrict__ pooled, const float* __restrict__ rw,
                          const float* __restrict__ rb, const float* __restrict__ bexp,
                          float* __restrict__ r, float* __restrict__ bb,
                          int Cin, int Cout) {
    int b = blockIdx.x;
    __shared__ float rs[8];
    int t = threadIdx.x;                       // blockDim = 64
    if (t < 8) {
        float a = rb[t];
        const float* p = pooled + (long long)b * Cin;
        for (int c = 0; c < Cin; ++c) a += p[c] * rw[c * 8 + t];
        float rv = 1.f / (1.f + __expf(-a));
        rs[t] = rv;
        r[b * 8 + t] = rv;
    }
    __syncthreads();
    for (int co = t; co < Cout; co += 64) {
        float a = 0.f;
        #pragma unroll
        for (int e = 0; e < 8; ++e) a += rs[e] * bexp[e * Cout + co];
        bb[(long long)b * Cout + co] = a;
    }
}

// ---------------------------------------------------------------------------
// Per-sample expert mixing: one block per (m,b) row; coalesced along kp.
// mw[(b*Cout+m)*Kp + kp] = sum_e r[b,e]*w[e,m,kp] (bf16, K zero-padded)
__global__ void k_mix(const float* __restrict__ w, const float* __restrict__ r,
                      u16* __restrict__ mw, int Cout, int Kdim, int Kp) {
    int m = blockIdx.x, b = blockIdx.y;
    float rl[8];
    #pragma unroll
    for (int e = 0; e < 8; ++e) rl[e] = r[b * 8 + e];
    const float* wp = w + (long long)m * Kdim;
    long long es = (long long)Cout * Kdim;
    u16* dst = mw + ((long long)b * Cout + m) * Kp;
    for (int kp = threadIdx.x; kp < Kp; kp += blockDim.x) {
        float v = 0.f;
        if (kp < Kdim) {
            #pragma unroll
            for (int e = 0; e < 8; ++e) v += rl[e] * wp[(long long)e * es + kp];
        }
        dst[kp] = f2bf(v);
    }
}

// ---------------------------------------------------------------------------
// One K-chunk of the implicit-GEMM conv: stage im2col B tile, load A frags, WMMA.
// FULL=true  -> chunk entirely inside Kdim (no K guard).
// interior   -> thread's whole kh x kw window is in-bounds (no spatial guard).
template<int KW, int MW, bool FULL>
__device__ __forceinline__ void conv_kstep(
    int k0, int Kdim, int Kp,
    const u16* __restrict__ inB, int H, int W,
    bool nvalid, bool interior, int ihBase, int iwBase,
    u16* __restrict__ ldsB, u16* __restrict__ ldsRow, int kkb,
    const u16* (&Arow)[MW], int aOff, int bOff, int laneM,
    v8f (&acc)[MW][4])
{
    constexpr int KHW = KW * KW;
    if (FULL && interior) {
        // fast path: no guards, straight-line loads
        #pragma unroll
        for (int j = 0; j < 16; j += 2) {
            u32 d = 0;
            #pragma unroll
            for (int h2 = 0; h2 < 2; ++h2) {
                int k = k0 + kkb + j + h2;
                int cin = k / KHW, rem = k - cin * KHW;   // compile-time magic div
                int rr = rem / KW,  ss = rem - rr * KW;
                float v = bf2f(inB[((long long)cin * H + ihBase + rr) * W + iwBase + ss]);
                d |= ((u32)f2bf(v)) << (16 * h2);
            }
            __builtin_memcpy(ldsRow + j, &d, 4);
        }
    } else {
        #pragma unroll
        for (int j = 0; j < 16; j += 2) {
            u32 d = 0;
            #pragma unroll
            for (int h2 = 0; h2 < 2; ++h2) {
                int k = k0 + kkb + j + h2;
                float v = 0.f;
                if (nvalid && (FULL || k < Kdim)) {
                    int cin = k / KHW, rem = k - cin * KHW;
                    int rr = rem / KW,  ss = rem - rr * KW;
                    int ih = ihBase + rr, iw = iwBase + ss;
                    if (ih >= 0 && ih < H && iw >= 0 && iw < W)
                        v = bf2f(inB[((long long)cin * H + ih) * W + iw]);
                }
                d |= ((u32)f2bf(v)) << (16 * h2);
            }
            __builtin_memcpy(ldsRow + j, &d, 4);
        }
    }
    __syncthreads();

    // A fragments: two contiguous 16B runs per lane (K 0..7 / 8..15, +16 for hi half)
    union AF { uint4 q[2]; v16bf v; } af[MW];
    #pragma unroll
    for (int f = 0; f < MW; ++f) {
        const u16* ap = Arow[f] + k0 + aOff;
        __builtin_memcpy(&af[f].q[0], ap, 16);
        __builtin_memcpy(&af[f].q[1], ap + 8, 16);
        if (k0 + 32 < Kp) __builtin_prefetch(ap + 32, 0, 2);   // global_prefetch_b8
    }
    // WMMA sweep: 4 N-subtiles x MW M-fragments
    #pragma unroll
    for (int ns = 0; ns < 4; ++ns) {
        union BF { uint4 q[2]; v16bf v; } bfr;
        const u16* bp = &ldsB[(ns * 16 + laneM) * LDSK + bOff];
        __builtin_memcpy(&bfr.q[0], bp, 16);        // ds_load_b128
        __builtin_memcpy(&bfr.q[1], bp + 8, 16);    // ds_load_b128
        #pragma unroll
        for (int f = 0; f < MW; ++f)
            acc[f][ns] = __builtin_amdgcn_wmma_f32_16x16x32_bf16(
                false, af[f].v, false, bfr.v, (short)0, acc[f][ns], false, false);
    }
    __syncthreads();
}

// ---------------------------------------------------------------------------
// CondConv as per-sample implicit GEMM: D[Cout, OH*OW] = Wmix[Cout,Kp] x im2col + bias, ReLU.
// 128 threads = 4 waves stacked along M; each wave owns 16*MW M-rows x 64 N-cols.
template<int KW, int MW>
__global__ void __launch_bounds__(128) k_condconv_wmma(
    const u16* __restrict__ act, const u16* __restrict__ mw, const float* __restrict__ bb,
    u16* __restrict__ out,
    int Cin, int H, int W, int Cout, int OH, int OW,
    int stride, int pad, int Kdim, int Kp, int NT)
{
    __shared__ __align__(16) u16 ldsB[64 * LDSK];
    int b = blockIdx.y;
    int mt = blockIdx.x / NT, nt = blockIdx.x % NT;
    int tid = threadIdx.x, wave = tid >> 5, lane = tid & 31;
    int laneM = lane & 15, hi = lane >> 4;
    int OHW = OH * OW;
    int n0 = nt * 64;
    int mBase = mt * (64 * MW) + wave * (16 * MW);

    const u16* Arow[MW];
    #pragma unroll
    for (int f = 0; f < MW; ++f) {
        int m = mBase + f * 16 + laneM;
        if (m >= Cout) m = Cout - 1;            // clamp; stores are guarded
        Arow[f] = mw + ((long long)b * Cout + m) * Kp;
    }
    int aOff = hi * 16;   // A-frag K offset (contiguous k = 16h + s, matches B)
    int bOff = hi * 16;   // B-frag K offset (contiguous k = 16h + s, matches A)

    v8f acc[MW][4] = {};

    // im2col staging: thread -> row nloc=tid>>1, half kkb=(tid&1)*16
    int nloc = tid >> 1, kkb = (tid & 1) * 16;
    int n = n0 + nloc;
    bool nvalid = n < OHW;
    int oh = nvalid ? n / OW : 0, ow = nvalid ? n % OW : 0;
    int ihBase = oh * stride - pad, iwBase = ow * stride - pad;
    bool interior = nvalid && ihBase >= 0 && (ihBase + KW) <= H
                           && iwBase >= 0 && (iwBase + KW) <= W;
    const u16* inB = act + (long long)b * Cin * H * W;
    u16* ldsRow = &ldsB[nloc * LDSK + kkb];

    int KMAIN = (Kdim >> 5) << 5;     // full chunks (== Kdim for layers 2..5)
    int k0 = 0;
    for (; k0 < KMAIN; k0 += 32)
        conv_kstep<KW, MW, true>(k0, Kdim, Kp, inB, H, W, nvalid, interior,
                                 ihBase, iwBase, ldsB, ldsRow, kkb,
                                 Arow, aOff, bOff, laneM, acc);
    for (; k0 < Kp; k0 += 32)         // guarded tail (layer 1 only)
        conv_kstep<KW, MW, false>(k0, Kdim, Kp, inB, H, W, nvalid, interior,
                                  ihBase, iwBase, ldsB, ldsRow, kkb,
                                  Arow, aOff, bOff, laneM, acc);

    // ---- epilogue: bias + ReLU, store bf16 activations
    const float* bbb = bb + (long long)b * Cout;
    #pragma unroll
    for (int f = 0; f < MW; ++f) {
        #pragma unroll
        for (int ns = 0; ns < 4; ++ns) {
            int nn = n0 + ns * 16 + laneM;
            #pragma unroll
            for (int v = 0; v < 8; ++v) {
                int m = mBase + f * 16 + v + hi * 8;
                if (nn < OHW && m < Cout) {
                    float val = acc[f][ns][v] + bbb[m];
                    val = val > 0.f ? val : 0.f;
                    out[((long long)b * Cout + m) * OHW + nn] = f2bf(val);
                }
            }
        }
    }
}

// ---------------------------------------------------------------------------
// 3x3 stride-2 max pool (VALID), bf16
__global__ void k_maxpool3s2(const u16* __restrict__ in, u16* __restrict__ out,
                             int C, int H, int W, int OH, int OW, long long total) {
    long long i = (long long)blockIdx.x * blockDim.x + threadIdx.x;
    if (i >= total) return;
    int ow = (int)(i % OW); long long t = i / OW;
    int oh = (int)(t % OH); t /= OH;
    int c  = (int)(t % C);  int b = (int)(t / C);
    const u16* p = in + (((long long)b * C + c) * H + oh * 2) * W + ow * 2;
    float m = -3.4e38f;
    #pragma unroll
    for (int r = 0; r < 3; ++r)
        #pragma unroll
        for (int s = 0; s < 3; ++s) {
            float v = bf2f(p[r * W + s]);
            m = v > m ? v : m;
        }
    out[i] = f2bf(m);
}

// ---------------------------------------------------------------------------
// FC GEMM: out[M=128,N] = act(bf16)[M,K] x wgt(f32->bf16)[K,N] + bias.
// GUARD: N not a multiple of 64 (fc3). RELU/OUTF32 compile-time.
template<bool GUARD, bool RELU, bool OUTF32>
__global__ void __launch_bounds__(128) k_fc_wmma(
    const u16* __restrict__ act, const float* __restrict__ wgt, const float* __restrict__ bias,
    void* __restrict__ outv, int N, int K, int NT)
{
    __shared__ __align__(16) u16 ldsB[64 * LDSK];
    int mt = blockIdx.x / NT, nt = blockIdx.x % NT;
    int tid = threadIdx.x, wave = tid >> 5, lane = tid & 31;
    int laneM = lane & 15, hi = lane >> 4;
    int n0 = nt * 64, mWave = mt * 64 + wave * 16;
    const u16* Arow = act + (long long)(mWave + laneM) * K;
    int aOff = hi * 16, bOff = hi * 16;
    v8f acc[4] = {};

    int nloc = tid >> 1, kkb = (tid & 1) * 16;
    int ncol = n0 + nloc;
    bool nvalid = !GUARD || (ncol < N);
    u16* ldsRow = &ldsB[nloc * LDSK + kkb];

    for (int k0 = 0; k0 < K; k0 += 32) {
        #pragma unroll
        for (int j = 0; j < 16; j += 2) {
            float v0, v1;
            if (GUARD) {
                v0 = nvalid ? wgt[(long long)(k0 + kkb + j)     * N + ncol] : 0.f;
                v1 = nvalid ? wgt[(long long)(k0 + kkb + j + 1) * N + ncol] : 0.f;
            } else {
                v0 = wgt[(long long)(k0 + kkb + j)     * N + ncol];
                v1 = wgt[(long long)(k0 + kkb + j + 1) * N + ncol];
            }
            u32 d = (u32)f2bf(v0) | ((u32)f2bf(v1) << 16);
            __builtin_memcpy(ldsRow + j, &d, 4);
        }
        if (nvalid && k0 + 32 < K)
            __builtin_prefetch(&wgt[(long long)(k0 + 32 + kkb) * N + ncol], 0, 2);
        __syncthreads();

        union AF { uint4 q[2]; v16bf v; } af;
        const u16* ap = Arow + k0 + aOff;
        __builtin_memcpy(&af.q[0], ap, 16);
        __builtin_memcpy(&af.q[1], ap + 8, 16);
        #pragma unroll
        for (int ns = 0; ns < 4; ++ns) {
            union BF { uint4 q[2]; v16bf v; } bfr;
            const u16* bp = &ldsB[(ns * 16 + laneM) * LDSK + bOff];
            __builtin_memcpy(&bfr.q[0], bp, 16);
            __builtin_memcpy(&bfr.q[1], bp + 8, 16);
            acc[ns] = __builtin_amdgcn_wmma_f32_16x16x32_bf16(
                false, af.v, false, bfr.v, (short)0, acc[ns], false, false);
        }
        __syncthreads();
    }

    #pragma unroll
    for (int ns = 0; ns < 4; ++ns) {
        int nn = n0 + ns * 16 + laneM;
        if (!GUARD || nn < N) {
            float bv = bias[nn];
            #pragma unroll
            for (int v = 0; v < 8; ++v) {
                int m = mWave + v + hi * 8;
                float val = acc[ns][v] + bv;
                if (RELU) val = val > 0.f ? val : 0.f;
                if (OUTF32) ((float*)outv)[(long long)m * N + nn] = val;
                else        ((u16*)  outv)[(long long)m * N + nn] = f2bf(val);
            }
        }
    }
}

// ---------------------------------------------------------------------------
// Host-side per-layer launcher (templated to pick the conv instantiation)
template<int KW, int MW>
static void launch_condconv(const u16* in, u16* out,
                            const float* W, const float* Bexp, const float* RW, const float* RB,
                            u16* mixed, float* pooled, float* rbuf, float* bbuf,
                            int Cin, int H, int Wd, int Cout, int strd, int pd,
                            int OH, int OW, int Kdim, int Kp, hipStream_t stream) {
    int HW = H * Wd;
    k_mean_hw<<<dim3(128u * Cin), dim3(256), 0, stream>>>(in, pooled, Cin, HW);
    k_routing<<<dim3(128), dim3(64), 0, stream>>>(pooled, RW, RB, Bexp, rbuf, bbuf, Cin, Cout);
    k_mix<<<dim3((unsigned)Cout, 128), dim3(256), 0, stream>>>(W, rbuf, mixed, Cout, Kdim, Kp);
    int OHW = OH * OW, NT = (OHW + 63) / 64;
    int MT = (Cout + 64 * MW - 1) / (64 * MW);
    k_condconv_wmma<KW, MW><<<dim3((unsigned)(MT * NT), 128), dim3(128), 0, stream>>>(
        in, mixed, bbuf, out, Cin, H, Wd, Cout, OH, OW, strd, pd, Kdim, Kp, NT);
}

// ---------------------------------------------------------------------------
extern "C" void kernel_launch(void* const* d_in, const int* in_sizes, int n_in,
                              void* d_out, int out_size, void* d_ws, size_t ws_size,
                              hipStream_t stream) {
    (void)in_sizes; (void)n_in; (void)out_size; (void)ws_size;
    const float* x = (const float*)d_in[0];
    const float *W[5], *Bexp[5], *RW[5], *RB[5];
    for (int l = 0; l < 5; ++l) {
        W[l]    = (const float*)d_in[1 + 4 * l];
        Bexp[l] = (const float*)d_in[2 + 4 * l];
        RW[l]   = (const float*)d_in[3 + 4 * l];
        RB[l]   = (const float*)d_in[4 + 4 * l];
    }
    const float *FW[3], *FB[3];
    for (int l = 0; l < 3; ++l) {
        FW[l] = (const float*)d_in[21 + 2 * l];
        FB[l] = (const float*)d_in[22 + 2 * l];
    }

    // Workspace layout
    char* ws = (char*)d_ws;
    size_t off = 0;
    auto take = [&](size_t bytes) -> char* {
        char* p = ws + off;
        off += (bytes + 255) & ~(size_t)255;
        return p;
    };
    const size_t ACT_CAP = 26214400ull;                 // max activation elems (L1 out = 24.78M)
    u16*   actA   = (u16*)take(ACT_CAP * 2);
    u16*   actB   = (u16*)take(ACT_CAP * 2);
    u16*   mixed  = (u16*)take(113246208ull * 2);       // L4 mixed weights: 128*256*3456 bf16
    float* pooled = (float*)take(128 * 384 * 4);
    float* rbuf   = (float*)take(128 * 8 * 4);
    float* bbuf   = (float*)take(128 * 384 * 4);

    // x -> bf16
    long long nx = (long long)128 * 3 * 224 * 224;
    k_f32_to_bf16<<<dim3((unsigned)((nx + 255) / 256)), dim3(256), 0, stream>>>(x, actA, nx);

    // Conv stack (KW, MW chosen so M-tiles fit Cout exactly: 64 / 192 / 2x192 / 2x128 / 2x128)
    launch_condconv<11, 1>(actA, actB, W[0], Bexp[0], RW[0], RB[0], mixed, pooled, rbuf, bbuf,
                           3, 224, 224, 64, 4, 2, 55, 55, 363, 384, stream);
    { long long t = (long long)128 * 64 * 27 * 27;
      k_maxpool3s2<<<dim3((unsigned)((t + 255) / 256)), dim3(256), 0, stream>>>(actB, actA, 64, 55, 55, 27, 27, t); }

    launch_condconv<5, 3>(actA, actB, W[1], Bexp[1], RW[1], RB[1], mixed, pooled, rbuf, bbuf,
                          64, 27, 27, 192, 1, 2, 27, 27, 1600, 1600, stream);
    { long long t = (long long)128 * 192 * 13 * 13;
      k_maxpool3s2<<<dim3((unsigned)((t + 255) / 256)), dim3(256), 0, stream>>>(actB, actA, 192, 27, 27, 13, 13, t); }

    launch_condconv<3, 3>(actA, actB, W[2], Bexp[2], RW[2], RB[2], mixed, pooled, rbuf, bbuf,
                          192, 13, 13, 384, 1, 1, 13, 13, 1728, 1728, stream);
    launch_condconv<3, 2>(actB, actA, W[3], Bexp[3], RW[3], RB[3], mixed, pooled, rbuf, bbuf,
                          384, 13, 13, 256, 1, 1, 13, 13, 3456, 3456, stream);
    launch_condconv<3, 2>(actA, actB, W[4], Bexp[4], RW[4], RB[4], mixed, pooled, rbuf, bbuf,
                          256, 13, 13, 256, 1, 1, 13, 13, 2304, 2304, stream);
    { long long t = (long long)128 * 256 * 6 * 6;
      k_maxpool3s2<<<dim3((unsigned)((t + 255) / 256)), dim3(256), 0, stream>>>(actB, actA, 256, 13, 13, 6, 6, t); }

    // FC stack: M=128 (2 M-tiles of 64)
    k_fc_wmma<false, true,  false><<<dim3(2u * 64), dim3(128), 0, stream>>>(actA, FW[0], FB[0], actB, 4096, 9216, 64);
    k_fc_wmma<false, true,  false><<<dim3(2u * 64), dim3(128), 0, stream>>>(actB, FW[1], FB[1], actA, 4096, 4096, 64);
    k_fc_wmma<true,  false, true ><<<dim3(2u * 16), dim3(128), 0, stream>>>(actA, FW[2], FB[2], d_out, 1000, 4096, 16);
}
